// BilstmCRF_81612968559244
// MI455X (gfx1250) — matrix operand order repaired and checked by reading, not compiled, not run on hardware
//
#include <hip/hip_runtime.h>
#include <math.h>

typedef _Float16 v16h __attribute__((ext_vector_type(16)));
typedef _Float16 v8h  __attribute__((ext_vector_type(8)));
typedef float    v8f  __attribute__((ext_vector_type(8)));

// Problem constants
#define B_    64
#define T_    512
#define E_    256
#define H_    512
#define G4H   2048          // 4*H
#define KCAT  768           // E + H
#define NT_   12
#define C_    10
#define PADT  0
#define BOS_  10
#define EOS_  11
#define BLKS_PER_DIR 16

// LDS staging strides (halves / floats), padded so row-parallel fragment reads
// land on disjoint banks (stride % 64 words == 4).
#define HSTR  520           // h rows: 512 + 8 halves
#define FSTR  264           // fmap rows: 256 + 8 halves
#define GSTR  132           // gates rows: 128 + 4 floats
#define LDS_H_BYTES   (64 * HSTR * 2)        // 66560
#define LDS_F_BYTES   (64 * FSTR * 2)        // 33792
#define LDS_G_BYTES   (64 * GSTR * 4)        // 33792
#define LDS_TOTAL     (LDS_H_BYTES + 2 * LDS_F_BYTES + LDS_G_BYTES)  // 167936

// ---------------- WMMA fragment loaders (layouts per ISA 7.12.2) -------------
// A 16x32 f16: lane L -> row M=L%16, half=L/16; K = half*8+{0..7} and 16+half*8+{0..7}
__device__ inline v16h load_a_frag(const _Float16* row, int k0, int kh) {
    const v8h* p0 = reinterpret_cast<const v8h*>(row + k0 + kh * 8);
    const v8h* p1 = reinterpret_cast<const v8h*>(row + k0 + 16 + kh * 8);
    v8h lo = *p0, hi = *p1;
    v16h a;
#pragma unroll
    for (int j = 0; j < 8; ++j) { a[j] = lo[j]; a[8 + j] = hi[j]; }
    return a;
}
// B 32x16 f16 for D = A @ W^T: lane L -> W-row n=L%16, half=L/16; K = half*8+{0..7} and 16+half*8+{0..7}
__device__ inline v16h load_b_frag(const _Float16* wrow, int k0, int kh) {
    const v8h* p0 = reinterpret_cast<const v8h*>(wrow + k0 + kh * 8);
    const v8h* p1 = reinterpret_cast<const v8h*>(wrow + k0 + 16 + kh * 8);
    v8h lo = *p0, hi = *p1;
    v16h b;
#pragma unroll
    for (int j = 0; j < 8; ++j) { b[j] = lo[j]; b[8 + j] = hi[j]; }
    return b;
}

// ---------------- CDNA5 async global->LDS staging ----------------------------
__device__ inline void async_ld128(unsigned lds_off, const void* gaddr) {
    asm volatile("global_load_async_to_lds_b128 %0, %1, off"
                 :: "v"(lds_off), "v"((unsigned long long)(__SIZE_TYPE__)gaddr)
                 : "memory");
}
__device__ inline void wait_async_le8() { asm volatile("s_wait_asynccnt 0x8" ::: "memory"); }
__device__ inline void wait_async_le0() { asm volatile("s_wait_asynccnt 0x0" ::: "memory"); }

// ---------------- global (per-direction) barrier: monotonic counter ----------
__device__ inline void gbar(unsigned* cnt, unsigned target) {
    __syncthreads();
    if (threadIdx.x == 0) {
        __threadfence();
        atomicAdd(cnt, 1u);
        while (__hip_atomic_load(cnt, __ATOMIC_ACQUIRE, __HIP_MEMORY_SCOPE_AGENT) < target) {
            __builtin_amdgcn_s_sleep(1);
        }
    }
    __syncthreads();
    __threadfence();
}

// ---------------- prep kernels ----------------------------------------------
__global__ void init_kernel(const float* fc_w, _Float16* fcw16, _Float16* h16, unsigned* cnt) {
    int tid = blockIdx.x * blockDim.x + threadIdx.x;
    int stride = gridDim.x * blockDim.x;
    for (int i = tid; i < 16 * 2 * H_; i += stride) {        // fc_w padded to 16 rows
        int c = i / (2 * H_), k = i % (2 * H_);
        fcw16[i] = (c < NT_) ? (_Float16)fc_w[(size_t)c * (2 * H_) + k] : (_Float16)0.f;
    }
    for (int i = tid; i < 2 * B_ * H_; i += stride) h16[i] = (_Float16)0.f;
    if (tid < 2) cnt[tid] = 0u;
}

__global__ void embed_kernel(const int* x, const float* emb, _Float16* fmap16) {
    size_t total = (size_t)B_ * T_ * E_;
    for (size_t i = (size_t)blockIdx.x * blockDim.x + threadIdx.x; i < total;
         i += (size_t)gridDim.x * blockDim.x) {
        size_t bt = i / E_;
        int e = (int)(i % E_);
        int tok = x[bt];
        fmap16[i] = (_Float16)emb[(size_t)tok * E_ + e];
    }
}

__global__ void wcat_kernel(const float* wih_f, const float* whh_f,
                            const float* wih_b, const float* whh_b, _Float16* wcat16) {
    size_t total = (size_t)2 * G4H * KCAT;
    for (size_t i = (size_t)blockIdx.x * blockDim.x + threadIdx.x; i < total;
         i += (size_t)gridDim.x * blockDim.x) {
        int dir = (int)(i / ((size_t)G4H * KCAT));
        size_t r = i % ((size_t)G4H * KCAT);
        int n = (int)(r / KCAT), k = (int)(r % KCAT);
        const float* wih = dir ? wih_b : wih_f;
        const float* whh = dir ? whh_b : whh_f;
        float v = (k < E_) ? wih[(size_t)n * E_ + k] : whh[(size_t)n * H_ + (k - E_)];
        wcat16[i] = (_Float16)v;
    }
}

// ---------------- fused BiLSTM recurrence ------------------------------------
// 32 blocks x 256 threads: blocks [0,16)=forward, [16,32)=backward.
// Block `blk` owns h-slice n in [blk*32, blk*32+32): it computes all 4 gate
// groups for that slice (columns g*512 + blk*32 + {0..31}), keeps gates in LDS,
// and updates h/c locally. Weights live in registers for all 512 steps; the
// A matrix [x_t | h] is staged in LDS with async loads (x_t double-buffered).
__global__ __launch_bounds__(256) void lstm_kernel(
    const int* __restrict__ x, const _Float16* __restrict__ fmap16,
    const _Float16* __restrict__ wcat16,
    const float* __restrict__ bias_f, const float* __restrict__ bias_b,
    _Float16* h16, _Float16* hiddens16, float* hT, unsigned* cnt)
{
    extern __shared__ char smem[];
    _Float16* hA  = (_Float16*)smem;                                   // 64 x HSTR
    _Float16* fA0 = (_Float16*)(smem + LDS_H_BYTES);                   // 64 x FSTR
    _Float16* fA1 = (_Float16*)(smem + LDS_H_BYTES + LDS_F_BYTES);     // 64 x FSTR
    float*    gts = (float*)(smem + LDS_H_BYTES + 2 * LDS_F_BYTES);    // 64 x GSTR

    const int dir  = blockIdx.x / BLKS_PER_DIR;
    const int blk  = blockIdx.x % BLKS_PER_DIR;
    const int tid  = threadIdx.x;
    const int lane = tid & 31, wave = tid >> 5;
    const int kh   = lane >> 4;            // fragment half
    const int lm   = lane & 15;

    const _Float16* wcat = wcat16 + (size_t)dir * G4H * KCAT;
    const float*    bias = dir ? bias_b : bias_f;
    _Float16*       hd   = h16 + (size_t)dir * B_ * H_;
    unsigned*       myc  = cnt + dir;

    // GEMM column assignment: wave w -> gate g=w/2, sub-tile s=w&1.
    const int gate = wave >> 1, sub = wave & 1;
    const int col_global = gate * H_ + blk * 32 + sub * 16 + lm;
    const int colL = wave * 16 + lm;       // == gate*32 + sub*16 + lm in gts rows
    const float bcol = bias[col_global];

    // Persistent weight registers: 24 fragments x 8 VGPRs = 192 VGPRs.
    const _Float16* wrow = wcat + (size_t)col_global * KCAT;
    v16h breg[KCAT / 32];
#pragma unroll
    for (int kk = 0; kk < KCAT / 32; ++kk) breg[kk] = load_b_frag(wrow, kk * 32, kh);

    // Elementwise ownership: thread owns (b = tid/4, n_local = (tid%4)*8 .. +8).
    const int eb = tid >> 2;
    const int nlb = (tid & 3) * 8;
    float hprev[8], cprev[8];
#pragma unroll
    for (int j = 0; j < 8; ++j) { hprev[j] = 0.f; cprev[j] = 0.f; }

    // Prefetch x_t for step 0.
    {
        const int t0 = dir ? (T_ - 1) : 0;
#pragma unroll
        for (int i = 0; i < 8; ++i) {
            int chunk = i * 256 + tid;             // 2048 chunks of 16B
            int row = chunk >> 5, kc = chunk & 31;
            async_ld128((unsigned)(__SIZE_TYPE__)(fA0 + row * FSTR + kc * 8),
                        fmap16 + ((size_t)row * T_ + t0) * E_ + kc * 8);
        }
    }

    unsigned round = 0;
    for (int t = 0; t < T_; ++t) {
        const int tcur = dir ? (T_ - 1 - t) : t;
        _Float16* fcur = (t & 1) ? fA1 : fA0;

        // Stage carried h (ready: previous step's global barrier passed).
#pragma unroll
        for (int i = 0; i < 16; ++i) {
            int chunk = i * 256 + tid;             // 4096 chunks of 16B
            int row = chunk >> 6, kc = chunk & 63;
            async_ld128((unsigned)(__SIZE_TYPE__)(hA + row * HSTR + kc * 8),
                        hd + (size_t)row * H_ + kc * 8);
        }
        if (t + 1 < T_) {
            // Prefetch next step's x_t into the other buffer; leave it in flight.
            const int tnext = dir ? (T_ - 2 - t) : (t + 1);
            _Float16* fnxt = (t & 1) ? fA0 : fA1;
#pragma unroll
            for (int i = 0; i < 8; ++i) {
                int chunk = i * 256 + tid;
                int row = chunk >> 5, kc = chunk & 31;
                async_ld128((unsigned)(__SIZE_TYPE__)(fnxt + row * FSTR + kc * 8),
                            fmap16 + ((size_t)row * T_ + tnext) * E_ + kc * 8);
            }
            wait_async_le8();   // in-order: h + current x_t complete, prefetch pending
        } else {
            wait_async_le0();
        }
        __syncthreads();

        // ---- GEMM: 4 M-tiles x 24 k-steps, A from LDS, B from registers ----
        v8f acc[4];
#pragma unroll
        for (int mt = 0; mt < 4; ++mt)
#pragma unroll
            for (int r = 0; r < 8; ++r) acc[mt][r] = 0.f;
#pragma unroll
        for (int kk = 0; kk < KCAT / 32; ++kk) {
            const int k0 = kk * 32;
#pragma unroll
            for (int mt = 0; mt < 4; ++mt) {
                const int row = mt * 16 + lm;
                v16h af = (k0 < E_)
                    ? load_a_frag(fcur + row * FSTR, k0, kh)
                    : load_a_frag(hA + row * HSTR, k0 - E_, kh);
                acc[mt] = __builtin_amdgcn_wmma_f32_16x16x32_f16(
                    false, af, false, breg[kk], (short)0, acc[mt], false, false);
            }
        }
        // D layout: c[r] -> M = r + 8*(lane/16); write gates to LDS
#pragma unroll
        for (int mt = 0; mt < 4; ++mt)
#pragma unroll
            for (int r = 0; r < 8; ++r) {
                const int m = mt * 16 + r + kh * 8;
                gts[m * GSTR + colL] = acc[mt][r] + bcol;
            }
        __syncthreads();

        // ---- gate nonlinearity + state update (block-local slice) ----
        const float mk = (x[(size_t)eb * T_ + tcur] != PADT) ? 1.f : 0.f;
#pragma unroll
        for (int j = 0; j < 8; ++j) {
            const int nl = nlb + j;
            const float gi = gts[eb * GSTR + nl];
            const float gf = gts[eb * GSTR + 32 + nl];
            const float gg = gts[eb * GSTR + 64 + nl];
            const float go = gts[eb * GSTR + 96 + nl];
            const float i_ = 1.f / (1.f + __expf(-gi));
            const float f_ = 1.f / (1.f + __expf(-gf));
            const float o_ = 1.f / (1.f + __expf(-go));
            const float g_ = tanhf(gg);
            const float cn = f_ * cprev[j] + i_ * g_;
            const float hn = o_ * tanhf(cn);
            const float hc = mk * hn + (1.f - mk) * hprev[j];
            const float cc = mk * cn + (1.f - mk) * cprev[j];
            hprev[j] = hc; cprev[j] = cc;
            const int ng = blk * 32 + nl;
            hd[(size_t)eb * H_ + ng] = (_Float16)hc;
            hiddens16[((size_t)eb * T_ + tcur) * (2 * H_) + dir * H_ + ng] = (_Float16)(mk * hn);
        }
        __builtin_amdgcn_s_cluster_barrier();   // NOP when unclustered
        gbar(myc, BLKS_PER_DIR * (++round));    // one device barrier per step
    }
#pragma unroll
    for (int j = 0; j < 8; ++j)
        hT[(size_t)dir * B_ * H_ + (size_t)eb * H_ + blk * 32 + nlb + j] = hprev[j];
}

// ---------------- emissions: (B*T,1024) @ fc_w^T via WMMA --------------------
__global__ __launch_bounds__(256) void emis_kernel(
    const _Float16* __restrict__ hiddens16, const _Float16* __restrict__ fcw16,
    const float* __restrict__ fc_b, float* emis)
{
    const int lane = threadIdx.x & 31, wave = threadIdx.x >> 5;
    const int kh = lane >> 4, lm = lane & 15;
    const int mtile = blockIdx.x * 8 + wave;       // 0..2047
    const int row0 = mtile * 16;
    const _Float16* A  = hiddens16 + ((size_t)row0 + lm) * (2 * H_);
    const _Float16* Bw = fcw16 + (size_t)lm * (2 * H_);

    v8f acc;
#pragma unroll
    for (int r = 0; r < 8; ++r) acc[r] = 0.f;
    for (int kk = 0; kk < (2 * H_) / 32; ++kk) {
        const int k0 = kk * 32;
        v16h af = load_a_frag(A, k0, kh);
        v16h bf = load_b_frag(Bw, k0, kh);
        acc = __builtin_amdgcn_wmma_f32_16x16x32_f16(false, af, false, bf,
                                                     (short)0, acc, false, false);
    }
    if (lm < NT_) {
        const float bb = fc_b[lm];
#pragma unroll
        for (int r = 0; r < 8; ++r) {
            const int m = row0 + r + kh * 8;
            emis[(size_t)m * NT_ + lm] = acc[r] + bb;
        }
    }
}

// ---------------- classifier head + softmax ----------------------------------
__global__ __launch_bounds__(256) void cls_kernel(
    const float* __restrict__ hT, const float* __restrict__ fc2_w,
    const float* __restrict__ fc2_b, float* out)
{
    __shared__ float logits[B_][C_];
    const float* hTf = hT;
    const float* hTb = hT + B_ * H_;
    for (int idx = threadIdx.x; idx < B_ * C_; idx += blockDim.x) {
        const int b = idx / C_, c = idx % C_;
        const float* w = fc2_w + (size_t)c * (2 * H_);
        float s = fc2_b[c];
        for (int k = 0; k < H_; ++k) s += hTf[(size_t)b * H_ + k] * w[k];
        for (int k = 0; k < H_; ++k) s += hTb[(size_t)b * H_ + k] * w[H_ + k];
        logits[b][c] = s;
    }
    __syncthreads();
    for (int b = threadIdx.x; b < B_; b += blockDim.x) {
        float mx = logits[b][0];
        for (int c = 1; c < C_; ++c) mx = fmaxf(mx, logits[b][c]);
        float e[C_], sum = 0.f;
        for (int c = 0; c < C_; ++c) { e[c] = __expf(logits[b][c] - mx); sum += e[c]; }
        for (int c = 0; c < C_; ++c) out[(size_t)b * C_ + c] = e[c] / sum;
    }
}

// ---------------- Viterbi (one wave32 per batch row) -------------------------
__global__ __launch_bounds__(32) void viterbi_kernel(
    const int* __restrict__ x, const float* __restrict__ emis,
    const float* __restrict__ trans, int* bpb, float* out_score, float* out_path)
{
    const int b = blockIdx.x, j = threadIdx.x;
    __shared__ float tr[NT_ * NT_];
    __shared__ float a_sh[NT_], a_nw[NT_];
    for (int i = j; i < NT_ * NT_; i += 32) tr[i] = trans[i];
    if (j < NT_) a_sh[j] = trans[BOS_ * NT_ + j] + emis[(size_t)b * T_ * NT_ + j];
    __syncthreads();
    for (int t = 1; t < T_; ++t) {
        const float mk = (x[(size_t)b * T_ + t] != PADT) ? 1.f : 0.f;
        if (j < NT_) {
            float best = -3.4e38f; int bi = 0;
            for (int i = 0; i < NT_; ++i) {
                const float s = a_sh[i] + tr[i * NT_ + j];
                if (s > best) { best = s; bi = i; }
            }
            const float e = emis[((size_t)b * T_ + t) * NT_ + j];
            const float an = (mk > 0.f) ? best + e : a_sh[j];
            const int   bp = (mk > 0.f) ? bi : j;
            a_nw[j] = an;
            bpb[((size_t)b * T_ + t) * NT_ + j] = bp;
        }
        __syncthreads();
        if (j < NT_) a_sh[j] = a_nw[j];
        __syncthreads();
    }
    if (j == 0) {
        float best = -3.4e38f; int bi = 0;
        for (int i = 0; i < NT_; ++i) {
            const float s = a_sh[i] + tr[i * NT_ + EOS_];
            if (s > best) { best = s; bi = i; }
        }
        out_score[b] = best;
        int tag = bi;
        for (int t = T_ - 1; t >= 1; --t) {
            const float mk = (x[(size_t)b * T_ + t] != PADT) ? 1.f : 0.f;
            out_path[(size_t)b * T_ + t] = (mk > 0.f) ? (float)tag : (float)PADT;
            tag = bpb[((size_t)b * T_ + t) * NT_ + tag];
        }
        const float m0 = (x[(size_t)b * T_] != PADT) ? 1.f : 0.f;
        out_path[(size_t)b * T_] = (m0 > 0.f) ? (float)tag : (float)PADT;
    }
}

// ---------------- host launch -------------------------------------------------
extern "C" void kernel_launch(void* const* d_in, const int* in_sizes, int n_in,
                              void* d_out, int out_size, void* d_ws, size_t ws_size,
                              hipStream_t stream) {
    const int*   x      = (const int*)  d_in[0];
    const float* emb    = (const float*)d_in[1];
    const float* w_ih_f = (const float*)d_in[2];
    const float* w_hh_f = (const float*)d_in[3];
    const float* b_f    = (const float*)d_in[4];
    const float* w_ih_b = (const float*)d_in[5];
    const float* w_hh_b = (const float*)d_in[6];
    const float* b_b    = (const float*)d_in[7];
    const float* fc_w   = (const float*)d_in[8];
    const float* fc_b   = (const float*)d_in[9];
    const float* fc2_w  = (const float*)d_in[10];
    const float* fc2_b  = (const float*)d_in[11];
    const float* trans  = (const float*)d_in[12];
    float* out = (float*)d_out;

    size_t off = 0;
    auto alloc = [&](size_t bytes) -> char* {
        off = (off + 255) & ~(size_t)255;
        char* p = (char*)d_ws + off;
        off += bytes;
        return p;
    };
    _Float16* fmap16    = (_Float16*)alloc((size_t)B_ * T_ * E_ * 2);
    _Float16* wcat16    = (_Float16*)alloc((size_t)2 * G4H * KCAT * 2);
    _Float16* h16       = (_Float16*)alloc((size_t)2 * B_ * H_ * 2);
    _Float16* hiddens16 = (_Float16*)alloc((size_t)B_ * T_ * 2 * H_ * 2);
    float*    hT        = (float*)   alloc((size_t)2 * B_ * H_ * 4);
    _Float16* fcw16     = (_Float16*)alloc((size_t)16 * 2 * H_ * 2);
    float*    emis      = (float*)   alloc((size_t)B_ * T_ * NT_ * 4);
    int*      bpb       = (int*)     alloc((size_t)B_ * T_ * NT_ * 4);
    unsigned* cnt       = (unsigned*)alloc(256);
    (void)in_sizes; (void)n_in; (void)out_size; (void)ws_size;

    init_kernel <<<64,  256, 0, stream>>>(fc_w, fcw16, h16, cnt);
    embed_kernel<<<1024,256, 0, stream>>>(x, emb, fmap16);
    wcat_kernel <<<1024,256, 0, stream>>>(w_ih_f, w_hh_f, w_ih_b, w_hh_b, wcat16);
    lstm_kernel <<<32,  256, LDS_TOTAL, stream>>>(x, fmap16, wcat16, b_f, b_b,
                                                  h16, hiddens16, hT, cnt);
    emis_kernel <<<256, 256, 0, stream>>>(hiddens16, fcw16, fc_b, emis);
    cls_kernel  <<<1,   256, 0, stream>>>(hT, fc2_w, fc2_b, out + B_ + B_ * T_);
    viterbi_kernel<<<B_, 32, 0, stream>>>(x, emis, trans, bpb, out, out + B_);
}
